// BMMAttention_25494925869081
// MI455X (gfx1250) — matrix operand order repaired and checked by reading, not compiled, not run on hardware
//
#include <hip/hip_runtime.h>
#include <hip/hip_bf16.h>
#include <stdint.h>

// ---------------------------------------------------------------------------
// Flash attention, B=16, S=2048, D=128 fp32 (q=k=v, unscaled softmax).
// bf16 WMMA (v_wmma_f32_16x16x32_bf16), fp32 online softmax + accumulators.
// Round 4: 6-arg tensor_load_to_lds (clang-23 arity) stages the K tile via
// TDM; Q pre-scaled by log2(e) -> raw v_exp_f32; v_perm_b32 transpose pack.
// ---------------------------------------------------------------------------

typedef __attribute__((ext_vector_type(16))) __bf16 v16bf;
typedef __attribute__((ext_vector_type(8)))  float  v8f;
typedef unsigned int u32x4 __attribute__((ext_vector_type(4)));
typedef int          i32x8 __attribute__((ext_vector_type(8)));
typedef int          i32x4 __attribute__((ext_vector_type(4)));

#define BATCH    16
#define SEQ      2048
#define DIM      128
#define QTILE    128        // q rows per workgroup: 8 waves x 16 rows
#define KVTILE   64         // keys per inner iteration
#define NWAVE    8
#define KSTRIDE  136        // bf16/row, K tile: 256B data + 16B TDM pad
#define VTSTRIDE 72         // bf16/row, transposed V tile (64 + 8 pad)
#define PSTRIDE  72         // bf16/row, per-wave P scratch (64 + 8 pad)

__device__ __forceinline__ uint16_t f32_to_bf16(float f) {
  union { float f; uint32_t u; } x; x.f = f;
  uint32_t u = x.u;
  u += 0x7fffu + ((u >> 16) & 1u);   // RNE
  return (uint16_t)(u >> 16);
}

__device__ __forceinline__ v8f wmma_bf16(v16bf a, v16bf b, v8f c) {
  return __builtin_amdgcn_wmma_f32_16x16x32_bf16(false, a, false, b,
                                                 (short)0, c, false, false);
}

__device__ __forceinline__ float fexp2(float x) {
  return __builtin_amdgcn_exp2f(x);      // raw v_exp_f32
}

// DPP16 row rotate: lanes 0-15 / 16-31 are separate DPP rows == our halves.
template <int CTRL>
__device__ __forceinline__ float dpp_ror(float x) {
  union { float f; int i; } u; u.f = x;
  u.i = __builtin_amdgcn_update_dpp(u.i, u.i, CTRL, 0xF, 0xF, true);
  return u.f;
}
__device__ __forceinline__ float rowmax16(float x) {
  x = fmaxf(x, dpp_ror<0x121>(x));
  x = fmaxf(x, dpp_ror<0x122>(x));
  x = fmaxf(x, dpp_ror<0x124>(x));
  x = fmaxf(x, dpp_ror<0x128>(x));
  return x;
}
__device__ __forceinline__ float rowsum16(float x) {
  x += dpp_ror<0x121>(x);
  x += dpp_ror<0x122>(x);
  x += dpp_ror<0x124>(x);
  x += dpp_ror<0x128>(x);
  return x;
}

// ------------------------- fp32 -> bf16 conversion -------------------------
__global__ __launch_bounds__(256)
void cvt_f32_bf16_kernel(const float* __restrict__ in,
                         uint16_t* __restrict__ out, int n4) {
  int i = blockIdx.x * 256 + threadIdx.x;
  if (i < n4) {
    float4 f = ((const float4*)in)[i];
    ushort4 o;
    o.x = f32_to_bf16(f.x);
    o.y = f32_to_bf16(f.y);
    o.z = f32_to_bf16(f.z);
    o.w = f32_to_bf16(f.w);
    ((ushort4*)out)[i] = o;
  }
}

// ----------------------------- flash attention -----------------------------
__global__ __launch_bounds__(256)
void fa_bf16_wmma_kernel(const uint16_t* __restrict__ kvq,
                         float* __restrict__ out) {
  __shared__ uint16_t sK [KVTILE * KSTRIDE];       // [key][d], filled by TDM
  __shared__ uint16_t sVT[DIM * VTSTRIDE];         // [d][key] transposed
  __shared__ uint16_t sP [NWAVE * 16 * PSTRIDE];   // per-wave P relayout

  const int tid  = threadIdx.x;
  const int lane = tid & 31;
  const int wave = tid >> 5;
  const int half = lane >> 4;
  const int l16  = lane & 15;

  const int bIdx = blockIdx.x >> 4;        // 16 q-tiles per batch
  const int q0   = (blockIdx.x & 15) * QTILE;
  const uint64_t base = (uint64_t)bIdx * SEQ * DIM;

  // ---- Q fragment (A-layout, in VGPRs), pre-scaled by log2(e) ----
  v16bf qfrag[4];
  {
    const int qrow = q0 + wave * 16 + l16;
    const uint16_t* qp = kvq + base + (uint64_t)qrow * DIM + half * 8;
    for (int c = 0; c < 4; ++c) {
      const uint16_t* p = qp + c * 32;
      ((uint4*)&qfrag[c])[0] = *(const uint4*)(p);
      ((uint4*)&qfrag[c])[1] = *(const uint4*)(p + 16);
    }
    #pragma unroll
    for (int c = 0; c < 4; ++c)
      #pragma unroll
      for (int i = 0; i < 16; ++i)
        qfrag[c][i] = (__bf16)((float)qfrag[c][i] * 1.44269504f);
  }

  v8f acc[8];
  float mrun[8], lrun[8];
  for (int t = 0; t < 8; ++t)
    for (int v = 0; v < 8; ++v) acc[t][v] = 0.0f;
  for (int v = 0; v < 8; ++v) { mrun[v] = -__builtin_inff(); lrun[v] = 0.0f; }

  for (int kv0 = 0; kv0 < SEQ; kv0 += KVTILE) {
    // ---- TDM: DMA the 64x128 bf16 tile into sK with padded rows ----
    if (wave == 0) {
      const uint64_t ga =
          (uint64_t)(uintptr_t)(kvq + base + (uint64_t)kv0 * DIM);
      u32x4 g0;
      g0[0] = 1u;                                   // count=1, user mode
      g0[1] = (uint32_t)(uintptr_t)&sK[0];          // lds_addr (bytes)
      g0[2] = (uint32_t)ga;                         // global_addr[31:0]
      g0[3] = (uint32_t)((ga >> 32) & 0x01FFFFFFu)  // global_addr[56:32]
              | (2u << 30);                         // type = 2 (image)
      i32x8 g1;
      g1[0] = (int)((1u << 16)        // data_size = 2 bytes
                  | (1u << 20)        // pad_enable
                  | (5u << 22)        // pad_interval: 64 DW = 256 B
                  | (3u << 25));      // pad_amount: 4 DW = 16 B
      g1[1] = (int)(DIM << 16);       // tensor_dim0 = 128 (lo16)
      g1[2] = (int)(KVTILE << 16);    // tensor_dim1 = 64  (lo16)
      g1[3] = (int)(DIM << 16);       // tile_dim0 = 128
      g1[4] = (int)KVTILE;            // tile_dim1 = 64, tile_dim2 = 0
      g1[5] = (int)DIM;               // tensor_dim0_stride = 128
      g1[6] = 0;
      g1[7] = 0;
      i32x4 z4 = {0, 0, 0, 0};
      i32x8 z8 = {0, 0, 0, 0, 0, 0, 0, 0};
      __builtin_amdgcn_tensor_load_to_lds(g0, g1, z4, z4, z8, 0);
    }

    // ---- transposed copy: global loads -> v_perm pair pack -> sVT ----
    {
      const int kq   = tid >> 4;          // 0..15 -> 4 rows each
      const int dblk = (tid & 15) * 8;    // 8 consecutive d values
      #pragma unroll
      for (int m = 0; m < 2; ++m) {
        const int r0 = kq * 4 + m * 2;    // even row pair
        const uint16_t* g0 = kvq + base + (uint64_t)(kv0 + r0) * DIM + dblk;
        uint4 ra = *(const uint4*)(g0);
        uint4 rb = *(const uint4*)(g0 + DIM);
        const uint32_t* wa = (const uint32_t*)&ra;
        const uint32_t* wb = (const uint32_t*)&rb;
        #pragma unroll
        for (int w = 0; w < 4; ++w) {
          uint32_t lo = __builtin_amdgcn_perm(wb[w], wa[w], 0x05040100u);
          uint32_t hi = __builtin_amdgcn_perm(wb[w], wa[w], 0x07060302u);
          *(uint32_t*)&sVT[(dblk + 2 * w)     * VTSTRIDE + r0] = lo;
          *(uint32_t*)&sVT[(dblk + 2 * w + 1) * VTSTRIDE + r0] = hi;
        }
        if (kv0 + KVTILE < SEQ)
          __builtin_prefetch(g0 + (size_t)KVTILE * DIM, 0, 0);
      }
    }
    if (wave == 0) __builtin_amdgcn_s_wait_tensorcnt(0);
    __syncthreads();

    // ---- S = Q K^T : four 16x16 score tiles, K chained over D=128 ----
    v8f st[4];
    for (int t = 0; t < 4; ++t)
      for (int v = 0; v < 8; ++v) st[t][v] = 0.0f;
    #pragma unroll
    for (int c = 0; c < 4; ++c) {
      #pragma unroll
      for (int t = 0; t < 4; ++t) {
        v16bf bk;
        const uint16_t* kp = &sK[(t * 16 + l16) * KSTRIDE + c * 32 + half * 8];
        ((uint4*)&bk)[0] = *(const uint4*)(kp);
        ((uint4*)&bk)[1] = *(const uint4*)(kp + 16);
        st[t] = wmma_bf16(qfrag[c], bk, st[t]);
      }
    }

    // ---- online softmax in base-2 domain (Q carries log2e) ----
    float alpha[8];
    v8f pt[4];
    #pragma unroll
    for (int v = 0; v < 8; ++v) {
      float r = fmaxf(fmaxf(st[0][v], st[1][v]), fmaxf(st[2][v], st[3][v]));
      r = rowmax16(r);
      float mnew = fmaxf(mrun[v], r);
      alpha[v] = fexp2(mrun[v] - mnew);
      mrun[v] = mnew;
      float e0 = fexp2(st[0][v] - mnew);
      float e1 = fexp2(st[1][v] - mnew);
      float e2 = fexp2(st[2][v] - mnew);
      float e3 = fexp2(st[3][v] - mnew);
      pt[0][v] = e0; pt[1][v] = e1; pt[2][v] = e2; pt[3][v] = e3;
      float rs = rowsum16((e0 + e1) + (e2 + e3));
      lrun[v] = lrun[v] * alpha[v] + rs;
    }
    #pragma unroll
    for (int t = 0; t < 8; ++t)
      #pragma unroll
      for (int v = 0; v < 8; ++v) acc[t][v] *= alpha[v];

    // ---- relayout P (16x64): C-layout -> A-layout via per-wave LDS ----
    {
      uint16_t* pw = &sP[wave * 16 * PSTRIDE];
      #pragma unroll
      for (int t = 0; t < 4; ++t) {
        #pragma unroll
        for (int v = 0; v < 8; ++v) {
          const int qrow = v + half * 8;
          union { float f; uint32_t u; } x; x.f = pt[t][v];
          pw[qrow * PSTRIDE + t * 16 + l16] = (uint16_t)(x.u >> 16); // trunc
        }
      }
    }
    asm volatile("s_wait_dscnt 0" ::: "memory");   // wave-local LDS RAW
    v16bf pfrag[2];
    #pragma unroll
    for (int c = 0; c < 2; ++c) {
      const uint16_t* pr =
          &sP[wave * 16 * PSTRIDE + l16 * PSTRIDE + c * 32 + half * 8];
      ((uint4*)&pfrag[c])[0] = *(const uint4*)(pr);
      ((uint4*)&pfrag[c])[1] = *(const uint4*)(pr + 16);
    }

    // ---- O += P V : 8 output tiles x 2 key chunks ----
    #pragma unroll
    for (int t = 0; t < 8; ++t) {
      #pragma unroll
      for (int c = 0; c < 2; ++c) {
        v16bf vfrag;
        const uint16_t* vr =
            &sVT[(t * 16 + l16) * VTSTRIDE + c * 32 + half * 8];
        ((uint4*)&vfrag)[0] = *(const uint4*)(vr);
        ((uint4*)&vfrag)[1] = *(const uint4*)(vr + 16);
        acc[t] = wmma_bf16(pfrag[c], vfrag, acc[t]);
      }
    }
    __syncthreads();   // before next tile overwrites sK/sVT
  }

  // ---- normalize and write fp32 output ----
  #pragma unroll
  for (int v = 0; v < 8; ++v) {
    const int row = q0 + wave * 16 + v + half * 8;
    const float inv = 1.0f / lrun[v];
    float* op = out + base + (uint64_t)row * DIM + l16;
    #pragma unroll
    for (int t = 0; t < 8; ++t) op[t * 16] = acc[t][v] * inv;
  }
}

// ------------------------------- launcher ---------------------------------
extern "C" void kernel_launch(void* const* d_in, const int* in_sizes, int n_in,
                              void* d_out, int out_size, void* d_ws, size_t ws_size,
                              hipStream_t stream) {
  const float* hs  = (const float*)d_in[0];
  float*       out = (float*)d_out;
  uint16_t*    bf  = (uint16_t*)d_ws;      // needs B*S*D*2 = 8.4 MB scratch

  const int total = in_sizes[0];           // 16*2048*128
  const int n4 = total / 4;
  hipLaunchKernelGGL(cvt_f32_bf16_kernel, dim3((n4 + 255) / 256), dim3(256),
                     0, stream, hs, bf, n4);
  hipLaunchKernelGGL(fa_bf16_wmma_kernel, dim3(BATCH * (SEQ / QTILE)),
                     dim3(256), 0, stream, bf, out);
}
